// CrossModelAtt_23673859735900
// MI455X (gfx1250) — matrix operand order repaired and checked by reading, not compiled, not run on hardware
//
#include <hip/hip_runtime.h>

// ---------------------------------------------------------------------------
// CrossModelAtt for MI455X (gfx1250, wave32, WMMA).
//   attn = softmax(Q KV^T); out = gamma * (attn KV) + img
// B=16, C=512, HW=4096. Both GEMMs on V_WMMA_F32_16X16X32_BF16, f32 accum.
// Workspace: attnF (f32, 16MB) | attnB (bf16, 8MB) | kvT (bf16, 64MB) = 88MB
// -> stays resident in the 192MB L2 between kernels.
// GEMMs: 128x128 block tile, BK=32, 8 waves (2x4), double-buffered LDS,
// one barrier per K-step, next tile's global loads overlap the WMMA burst.
// All dims are template constants so epilogue stores use immediate offsets.
// ---------------------------------------------------------------------------

typedef __bf16 bf16_t;
typedef __attribute__((ext_vector_type(16))) __bf16 v16bf;
typedef __attribute__((ext_vector_type(8)))  float  v8f;

union FragAB  { v16bf v; uint4 u[2]; };   // one WMMA A/B operand (32B)
union FragC   { v8f   v; float f[8]; };   // one 16x16 f32 accumulator (32B)
union Stage16 { bf16_t h[16]; uint4 u[2]; };  // 16 bf16 staged per thread

#define BM 128
#define BN 128
#define BK 32
#define LDT 40            // padded LDS row stride in bf16 (80B, 16B aligned)
#define TILE_E (BM * LDT) // 5120 elems = 10240 B per tile buffer

__device__ __forceinline__ void cvt_stage(const float* src, Stage16& s) {
  const float4* p = (const float4*)src;
  #pragma unroll
  for (int v = 0; v < 4; ++v) {
    float4 f = p[v];
    s.h[4*v+0] = (bf16_t)f.x; s.h[4*v+1] = (bf16_t)f.y;
    s.h[4*v+2] = (bf16_t)f.z; s.h[4*v+3] = (bf16_t)f.w;
  }
}

// ---------------------------------------------------------------------------
// GEMM1: C[b][m][n] = sum_k A[b][m][k] * Bm[b][n][k]   (NT, f32 in, bf16 WMMA)
// ---------------------------------------------------------------------------
template<int M, int N, int K>
__global__ __launch_bounds__(256) void gemm_nt_qk(
    const float* __restrict__ A, const float* __restrict__ Bm,
    float* __restrict__ C)
{
  __shared__ __align__(16) bf16_t As[2][TILE_E];
  __shared__ __align__(16) bf16_t Bs[2][TILE_E];

  const int b  = blockIdx.z;
  const int m0 = blockIdx.x * BM;
  const int n0 = blockIdx.y * BN;

  const int t    = threadIdx.x;
  const int lane = t & 31;
  const int wave = t >> 5;
  const int wm   = wave & 1;      // 2 wave-rows x 64 M
  const int wn   = wave >> 1;     // 4 wave-cols x 32 N
  const int srow = t >> 1;        // staging row 0..127
  const int skc  = (t & 1) * 16;  // staging K chunk

  // hoisted global pointers (advance by constant k0 only)
  const float* aptr = A + (size_t)b * M * K + (size_t)(m0 + srow) * K + skc;
  const float* bptr = Bm + (size_t)b * N * K + (size_t)(n0 + srow) * K + skc;

  // per-thread LDS addresses
  const int soff = srow * LDT + skc;
  const int aoff = (wm * 64 + (lane & 15)) * LDT + ((lane < 16) ? 0 : 8);
  const int boff = (wn * 32 + (lane & 15)) * LDT + ((lane < 16) ? 0 : 8);

  FragC acc[4][2] = {};

  Stage16 sa, sb;
  cvt_stage(aptr, sa);
  cvt_stage(bptr, sb);
  { uint4* da = (uint4*)&As[0][soff]; da[0] = sa.u[0]; da[1] = sa.u[1];
    uint4* db = (uint4*)&Bs[0][soff]; db[0] = sb.u[0]; db[1] = sb.u[1]; }

  int cur = 0;
  for (int k0 = 0; k0 < K; k0 += BK) {
    __syncthreads();
    const bool more = (k0 + BK) < K;
    if (more) {                      // issue next tile's global loads now;
      cvt_stage(aptr + k0 + BK, sa); // they retire under the WMMA burst
      cvt_stage(bptr + k0 + BK, sb);
    }

    FragAB bfr[2];
    #pragma unroll
    for (int j = 0; j < 2; ++j) {
      const uint4* p = (const uint4*)&Bs[cur][boff + j * 16 * LDT];
      bfr[j].u[0] = p[0]; bfr[j].u[1] = p[2];
    }
    #pragma unroll
    for (int i = 0; i < 4; ++i) {
      const bf16_t* pa = &As[cur][aoff + i * 16 * LDT];
      FragAB af;
      af.u[0] = *(const uint4*)pa;
      af.u[1] = *(const uint4*)(pa + 16);
      #pragma unroll
      for (int j = 0; j < 2; ++j)
        acc[i][j].v = __builtin_amdgcn_wmma_f32_16x16x32_bf16(
            false, af.v, false, bfr[j].v, (short)0, acc[i][j].v, false, false);
    }

    if (more) {
      cur ^= 1;
      uint4* da = (uint4*)&As[cur][soff]; da[0] = sa.u[0]; da[1] = sa.u[1];
      uint4* db = (uint4*)&Bs[cur][soff]; db[0] = sb.u[0]; db[1] = sb.u[1];
    }
  }

  // epilogue: VGPR e -> M=e (lanes 0-15) / M=e+8 (lanes 16-31), N=lane%16
  float* cb = C + (size_t)b * M * N;
  const int rbase = m0 + wm * 64 + ((lane >> 4) << 3);
  const int col   = n0 + wn * 32 + (lane & 15);
  #pragma unroll
  for (int i = 0; i < 4; ++i) {
    float* p = cb + (size_t)(rbase + i * 16) * N + col;
    #pragma unroll
    for (int j = 0; j < 2; ++j)
      #pragma unroll
      for (int e = 0; e < 8; ++e)
        p[e * N + j * 16] = acc[i][j].f[e];   // constant offsets -> immediates
  }
}

// ---------------------------------------------------------------------------
// GEMM2 + epilogue: out = gamma * (attnB kvT^T) + img   (bf16 NT WMMA)
// A:[B][M][K] bf16, Bm:[B][N][K] bf16, img/out:[B][M][N] f32
// ---------------------------------------------------------------------------
template<int M, int N, int K>
__global__ __launch_bounds__(256) void gemm_nt_av(
    const bf16_t* __restrict__ A, const bf16_t* __restrict__ Bm,
    const float* __restrict__ img, const float* __restrict__ gammap,
    float* __restrict__ out)
{
  __shared__ __align__(16) bf16_t As[2][TILE_E];
  __shared__ __align__(16) bf16_t Bs[2][TILE_E];

  const int b  = blockIdx.z;
  const int m0 = blockIdx.x * BM;
  const int n0 = blockIdx.y * BN;

  const int t    = threadIdx.x;
  const int lane = t & 31;
  const int wave = t >> 5;
  const int wm   = wave & 1;
  const int wn   = wave >> 1;
  const int srow = t >> 1;
  const int skc  = (t & 1) * 16;

  const bf16_t* aptr = A + (size_t)b * M * K + (size_t)(m0 + srow) * K + skc;
  const bf16_t* bptr = Bm + (size_t)b * N * K + (size_t)(n0 + srow) * K + skc;

  const int soff = srow * LDT + skc;
  const int aoff = (wm * 64 + (lane & 15)) * LDT + ((lane < 16) ? 0 : 8);
  const int boff = (wn * 32 + (lane & 15)) * LDT + ((lane < 16) ? 0 : 8);

  FragC acc[4][2] = {};

  Stage16 sa, sb;
  { const uint4* p = (const uint4*)aptr; sa.u[0] = p[0]; sa.u[1] = p[1];
    const uint4* q = (const uint4*)bptr; sb.u[0] = q[0]; sb.u[1] = q[1]; }
  { uint4* da = (uint4*)&As[0][soff]; da[0] = sa.u[0]; da[1] = sa.u[1];
    uint4* db = (uint4*)&Bs[0][soff]; db[0] = sb.u[0]; db[1] = sb.u[1]; }

  int cur = 0;
  for (int k0 = 0; k0 < K; k0 += BK) {
    __syncthreads();
    const bool more = (k0 + BK) < K;
    if (more) {
      const uint4* p = (const uint4*)(aptr + k0 + BK); sa.u[0] = p[0]; sa.u[1] = p[1];
      const uint4* q = (const uint4*)(bptr + k0 + BK); sb.u[0] = q[0]; sb.u[1] = q[1];
    }

    FragAB bfr[2];
    #pragma unroll
    for (int j = 0; j < 2; ++j) {
      const uint4* p = (const uint4*)&Bs[cur][boff + j * 16 * LDT];
      bfr[j].u[0] = p[0]; bfr[j].u[1] = p[2];
    }
    #pragma unroll
    for (int i = 0; i < 4; ++i) {
      const bf16_t* pa = &As[cur][aoff + i * 16 * LDT];
      FragAB af;
      af.u[0] = *(const uint4*)pa;
      af.u[1] = *(const uint4*)(pa + 16);
      #pragma unroll
      for (int j = 0; j < 2; ++j)
        acc[i][j].v = __builtin_amdgcn_wmma_f32_16x16x32_bf16(
            false, af.v, false, bfr[j].v, (short)0, acc[i][j].v, false, false);
    }

    if (more) {
      cur ^= 1;
      uint4* da = (uint4*)&As[cur][soff]; da[0] = sa.u[0]; da[1] = sa.u[1];
      uint4* db = (uint4*)&Bs[cur][soff]; db[0] = sb.u[0]; db[1] = sb.u[1];
    }
  }

  const float g = gammap[0];
  const float* ib = img + (size_t)b * M * N;
  float*       ob = out + (size_t)b * M * N;
  const int rbase = m0 + wm * 64 + ((lane >> 4) << 3);
  const int col   = n0 + wn * 32 + (lane & 15);
  #pragma unroll
  for (int i = 0; i < 4; ++i) {
    const size_t base = (size_t)(rbase + i * 16) * N + col;
    #pragma unroll
    for (int j = 0; j < 2; ++j)
      #pragma unroll
      for (int e = 0; e < 8; ++e) {
        const int off = e * N + j * 16;      // compile-time constant
        ob[base + off] = fmaf(g, acc[i][j].f[e], ib[base + off]);
      }
  }
}

// ---------------------------------------------------------------------------
// Row softmax over attn [B*C x 512] -> bf16. One wave32 per row.
// ---------------------------------------------------------------------------
__global__ __launch_bounds__(256) void softmax_rows(
    const float* __restrict__ attn, bf16_t* __restrict__ attnB)
{
  const int N    = 512;
  const int row  = blockIdx.x * 8 + (threadIdx.x >> 5);
  const int lane = threadIdx.x & 31;
  const float* p = attn + (size_t)row * N + lane * 16;

  float x[16];
  #pragma unroll
  for (int i = 0; i < 4; ++i) {
    float4 v = ((const float4*)p)[i];
    x[4*i+0] = v.x; x[4*i+1] = v.y; x[4*i+2] = v.z; x[4*i+3] = v.w;
  }
  float m = -3.402823466e38f;
  #pragma unroll
  for (int i = 0; i < 16; ++i) m = fmaxf(m, x[i]);
  #pragma unroll
  for (int off = 16; off >= 1; off >>= 1) m = fmaxf(m, __shfl_xor(m, off, 32));

  float s = 0.f;
  #pragma unroll
  for (int i = 0; i < 16; ++i) { x[i] = __expf(x[i] - m); s += x[i]; }
  #pragma unroll
  for (int off = 16; off >= 1; off >>= 1) s += __shfl_xor(s, off, 32);
  const float inv = 1.0f / s;

  Stage16 o;
  #pragma unroll
  for (int i = 0; i < 16; ++i) o.h[i] = (bf16_t)(x[i] * inv);
  uint4* q = (uint4*)(attnB + (size_t)row * N + lane * 16);
  q[0] = o.u[0]; q[1] = o.u[1];   // two global_store_b128
}

// ---------------------------------------------------------------------------
// kvT[b][n][d] = (bf16) text[b][d][n]
// ---------------------------------------------------------------------------
__global__ __launch_bounds__(256) void transpose_cvt(
    const float* __restrict__ src, bf16_t* __restrict__ dst)
{
  const int Cdim = 512, HW = 4096;
  __shared__ bf16_t tile[32][33];
  const int b  = blockIdx.z;
  const int c0 = blockIdx.y * 32;
  const int n0 = blockIdx.x * 32;
  const int tx = threadIdx.x & 31;
  const int ty = threadIdx.x >> 5;  // 0..7
  const float* s = src + (size_t)b * Cdim * HW;
  #pragma unroll
  for (int i = 0; i < 32; i += 8)
    tile[ty + i][tx] = (bf16_t)s[(size_t)(c0 + ty + i) * HW + n0 + tx];
  __syncthreads();
  bf16_t* d = dst + (size_t)b * HW * Cdim;
  #pragma unroll
  for (int i = 0; i < 32; i += 8)
    d[(size_t)(n0 + ty + i) * Cdim + c0 + tx] = tile[tx][ty + i];
}

// ---------------------------------------------------------------------------
extern "C" void kernel_launch(void* const* d_in, const int* in_sizes, int n_in,
                              void* d_out, int out_size, void* d_ws, size_t ws_size,
                              hipStream_t stream) {
  (void)in_sizes; (void)n_in; (void)out_size; (void)ws_size;
  const float* img   = (const float*)d_in[0];   // [16,512,64,64] f32
  const float* txt   = (const float*)d_in[1];   // [16,512,64,64] f32
  const float* gamma = (const float*)d_in[2];   // [1] f32
  float* out = (float*)d_out;

  constexpr int Bb = 16, Cc = 512, HW = 64 * 64;

  char* ws = (char*)d_ws;
  float*  attnF = (float*)ws;                                   // 16 MB
  bf16_t* attnB = (bf16_t*)(ws + (size_t)Bb * Cc * Cc * 4);     //  8 MB
  bf16_t* kvT   = (bf16_t*)(ws + (size_t)Bb * Cc * Cc * 6);     // 64 MB

  // 1) kvT = transpose(text) in bf16
  transpose_cvt<<<dim3(HW / 32, Cc / 32, Bb), 256, 0, stream>>>(txt, kvT);

  // 2) attnF = Q KV^T
  gemm_nt_qk<Cc, Cc, HW><<<dim3(Cc / BM, Cc / BN, Bb), 256, 0, stream>>>(
      img, txt, attnF);

  // 3) attnB = softmax_rows(attnF) in bf16
  softmax_rows<<<dim3(Bb * Cc / 8), 256, 0, stream>>>(attnF, attnB);

  // 4) out = gamma * (attnB kvT^T) + img
  gemm_nt_av<Cc, HW, Cc><<<dim3(Cc / BM, HW / BN, Bb), 256, 0, stream>>>(
      attnB, kvT, img, gamma, out);
}
